// SphericalFrequencyDomainSNN_27616639713862
// MI455X (gfx1250) — matrix operand order repaired and checked by reading, not compiled, not run on hardware
//
#include <hip/hip_runtime.h>
#include <math.h>

// ---------------------------------------------------------------------------
// SphericalFrequencyDomainSNN for MI455X (gfx1250, wave32, WMMA)
//
// Pipeline per timestep t (all GEMMs via V_WMMA_F32_16X16X32_F16):
//   1) gray      : mean over RGB -> f16 (A-matrix for row DFT)
//   2) row DFT   : R = G(224x224) x E(224x{113re,113im} pad 256), K=224=7x32
//                  output written TRANSPOSED (Rt[v][h]) so it is a contiguous
//                  B-operand for the column DFT.
//   3) col DFT   : S = F(224x224 cplx) x R ; complex via 4 WMMAs/k-step.
//                  CDNA5 f16 WMMA has no A/B negation, so
//                  the -Fim*Rim term uses a precomputed negated twiddle
//                  matrix Fin = -Fim. mag=log1p(|S|), energy accumulated.
//   4) upsample  : 113 -> 224 linear (half-pixel, edge clamp), pack f16
//                  feature pixels [r,g,b,mag] (8 B / pixel).
//   5) conv+LIF  : implicit GEMM, K=3*3*4=36 zero-padded to 64 (2 k-steps),
//                  16 pixels x 32 channels per wave (2 N-tiles), fused
//                  membrane update V=0.9V+cur, spike, soft reset, spike-sum
//                  reduction (shfl_xor 16 + atomics).
// Then one small head kernel builds logits/readout/spike-rate/energy.
// ---------------------------------------------------------------------------

typedef _Float16 h4  __attribute__((ext_vector_type(4)));
typedef _Float16 h8  __attribute__((ext_vector_type(8)));
typedef _Float16 v16h __attribute__((ext_vector_type(16)));
typedef float    v8f  __attribute__((ext_vector_type(8)));

#define T_N   16
#define B_N   8
#define H_N   224
#define W_N   224
#define CH_N  32
#define OUT_N 10
#define NFREQ 113
#define NF_PAD 128
#define K224  224
#define BETA_ 0.9f
#define THRESH_ 1.0f
#define TWO_PI 6.283185307179586f

// ---------------------------------------------------------------- utilities
__global__ void k_zero(float* __restrict__ p, size_t n) {
  size_t i = (size_t)blockIdx.x * 256 + threadIdx.x;
  if (i < n) p[i] = 0.0f;
}

// Wt[n][w]: transposed rFFT matrix. n<128 -> Re(e^{-2pi i w v/224}), v=n
//           n>=128 -> Im(...), v=n-128. Rows with v>=113 are zero.
__global__ void k_init_wt(_Float16* __restrict__ Wt) {
  int idx = blockIdx.x * 256 + threadIdx.x;
  if (idx >= 256 * K224) return;
  int w = idx % K224, n = idx / K224;
  int v = (n < 128) ? n : (n - 128);
  _Float16 val = (_Float16)0.0f;
  if (v < NFREQ) {
    float ph = -TWO_PI * (float)((w * v) % K224) / (float)K224; // exact mod
    val = (_Float16)((n < 128) ? cosf(ph) : sinf(ph));
  }
  Wt[(size_t)n * K224 + w] = val;
}

// Column-DFT matrices, row-major [u][h] (A-operand layout).
// Fre = cos(ph), Fim = sin(ph), Fin = -sin(ph) with ph = -2*pi*u*h/224.
// Fin exists because CDNA5 f16 WMMA has no A-negate modifier.
__global__ void k_init_f(_Float16* __restrict__ Fre, _Float16* __restrict__ Fim,
                         _Float16* __restrict__ Fin) {
  int idx = blockIdx.x * 256 + threadIdx.x;
  if (idx >= K224 * K224) return;
  int hh = idx % K224, u = idx / K224;
  float ph = -TWO_PI * (float)((u * hh) % K224) / (float)K224;
  float s = sinf(ph);
  Fre[idx] = (_Float16)cosf(ph);
  Fim[idx] = (_Float16)s;
  Fin[idx] = (_Float16)(-s);
}

// WcT[co][k] transposed conv weights, K padded 36 -> 64 with zeros.
// conv_w HWIO: flat index ((ky*3+kx)*4+ci)*32+co = k*32+co.
__global__ void k_init_wct(const float* __restrict__ conv_w, _Float16* __restrict__ WcT) {
  int idx = blockIdx.x * 256 + threadIdx.x;
  if (idx >= CH_N * 64) return;
  int k = idx % 64, co = idx / 64;
  _Float16 v = (_Float16)0.0f;
  if (k < 36) v = (_Float16)conv_w[(size_t)k * CH_N + co];
  WcT[(size_t)co * 64 + k] = v;
}

__global__ void k_gray(const float* __restrict__ x_t, _Float16* __restrict__ gray) {
  size_t idx = (size_t)blockIdx.x * 256 + threadIdx.x;
  if (idx >= (size_t)B_N * H_N * W_N) return;
  const float* p = x_t + idx * 3;
  gray[idx] = (_Float16)((p[0] + p[1] + p[2]) * (1.0f / 3.0f));
}

// ------------------------------------------------------------- row DFT GEMM
// One 16x16 tile per wave; 4 waves / 128-thread block.
// A = gray (row-major, M=h, K=w), B = Wt (pre-transposed, contiguous loads),
// D written transposed into Rt[v_combined][h].
__global__ void k_rowdft(const _Float16* __restrict__ gray,
                         const _Float16* __restrict__ Wt,
                         _Float16* __restrict__ Rt) {
  int tile = blockIdx.x * 4 + (threadIdx.x >> 5);
  int lane = threadIdx.x & 31;
  int img = tile / (14 * 16);
  int r   = tile % (14 * 16);
  int mt = r >> 4, nt = r & 15;
  int g = lane >> 4, ln = lane & 15;
  const _Float16* Ar = gray + (size_t)img * H_N * W_N + (size_t)(mt * 16 + ln) * W_N;
  const _Float16* Br = Wt + (size_t)(nt * 16 + ln) * K224;
  v8f acc = {};
#pragma unroll
  for (int k0 = 0; k0 < K224; k0 += 32) {
    h8 lo = *(const h8*)(Ar + k0 + 16 * g);
    h8 hi = *(const h8*)(Ar + k0 + 16 * g + 8);
    v16h a;
#pragma unroll
    for (int j = 0; j < 8; ++j) { a[j] = lo[j]; a[8 + j] = hi[j]; }
    v16h b = *(const v16h*)(Br + k0 + 16 * g);
    acc = __builtin_amdgcn_wmma_f32_16x16x32_f16(false, a, false, b, (short)0, acc, false, false);
  }
  // D fragment: lane owns column n=ln, rows m = mt*16 + i + 8*g -> write
  // transposed (Rt[n][m]) as one contiguous run of 8 f16.
  _Float16* out = Rt + (size_t)img * 256 * K224 + (size_t)(nt * 16 + ln) * K224 + mt * 16 + 8 * g;
#pragma unroll
  for (int i = 0; i < 8; ++i) out[i] = (_Float16)acc[i];
}

// ------------------------------------------------- column DFT (complex) GEMM
__global__ void k_coldft(const _Float16* __restrict__ Fre,
                         const _Float16* __restrict__ Fim,
                         const _Float16* __restrict__ Fin,
                         const _Float16* __restrict__ Rt,
                         float* __restrict__ mag,
                         float* __restrict__ energy) {
  int tile = blockIdx.x * 4 + (threadIdx.x >> 5);
  int lane = threadIdx.x & 31;
  int img = tile / (14 * 8);
  int r   = tile % (14 * 8);
  int ut = r >> 3, vt = r & 7;
  int g = lane >> 4, ln = lane & 15;
  const _Float16* Are = Fre + (size_t)(ut * 16 + ln) * K224;
  const _Float16* Aim = Fim + (size_t)(ut * 16 + ln) * K224;
  const _Float16* Ain = Fin + (size_t)(ut * 16 + ln) * K224;
  const _Float16* Bre = Rt + (size_t)img * 256 * K224 + (size_t)(vt * 16 + ln) * K224;
  const _Float16* Bim = Bre + (size_t)128 * K224;
  v8f ar = {}, ai = {};
#pragma unroll
  for (int k0 = 0; k0 < K224; k0 += 32) {
    h8 rl = *(const h8*)(Are + k0 + 16 * g);
    h8 rh = *(const h8*)(Are + k0 + 16 * g + 8);
    h8 il = *(const h8*)(Aim + k0 + 16 * g);
    h8 ih = *(const h8*)(Aim + k0 + 16 * g + 8);
    h8 nl = *(const h8*)(Ain + k0 + 16 * g);
    h8 nh = *(const h8*)(Ain + k0 + 16 * g + 8);
    v16h fre, fim, fin;
#pragma unroll
    for (int j = 0; j < 8; ++j) {
      fre[j] = rl[j]; fre[8 + j] = rh[j];
      fim[j] = il[j]; fim[8 + j] = ih[j];
      fin[j] = nl[j]; fin[8 + j] = nh[j];
    }
    v16h bre = *(const v16h*)(Bre + k0 + 16 * g);
    v16h bim = *(const v16h*)(Bim + k0 + 16 * g);
    // S_re += Fre*Rre + (-Fim)*Rim
    ar = __builtin_amdgcn_wmma_f32_16x16x32_f16(false, fre, false, bre, (short)0, ar, false, false);
    ar = __builtin_amdgcn_wmma_f32_16x16x32_f16(false, fin, false, bim, (short)0, ar, false, false);
    // S_im += Fre*Rim + Fim*Rre
    ai = __builtin_amdgcn_wmma_f32_16x16x32_f16(false, fre, false, bim, (short)0, ai, false, false);
    ai = __builtin_amdgcn_wmma_f32_16x16x32_f16(false, fim, false, bre, (short)0, ai, false, false);
  }
  int v = vt * 16 + ln;
  float esum = 0.0f;
  if (v < NFREQ) {
#pragma unroll
    for (int i = 0; i < 8; ++i) {
      int u = ut * 16 + i + 8 * g;
      float re = ar[i], im = ai[i];
      float val = log1pf(sqrtf(re * re + im * im));
      mag[((size_t)img * H_N + u) * NF_PAD + v] = val;
      esum += val;
    }
  }
#pragma unroll
  for (int off = 16; off > 0; off >>= 1) esum += __shfl_xor(esum, off, 32);
  if (lane == 0) atomicAdd(energy, esum);
}

// ------------------------------------------- feature pack (rgb + upsampled mag)
__global__ void k_feat(const float* __restrict__ x_t, const float* __restrict__ mag,
                       _Float16* __restrict__ feat) {
  size_t idx = (size_t)blockIdx.x * 256 + threadIdx.x;
  if (idx >= (size_t)B_N * H_N * W_N) return;
  int w = (int)(idx % W_N);
  size_t bh = idx / W_N;
  int h = (int)(bh % H_N);
  int b = (int)(bh / H_N);
  const float* px = x_t + idx * 3;
  // half-pixel linear resize 113 -> 224, edge clamp
  float src = ((float)w + 0.5f) * ((float)NFREQ / (float)W_N) - 0.5f;
  src = fmaxf(src, 0.0f);
  int i0 = (int)src; if (i0 > NFREQ - 1) i0 = NFREQ - 1;
  float f = src - (float)i0;
  int i1 = (i0 + 1 > NFREQ - 1) ? (NFREQ - 1) : (i0 + 1);
  const float* mrow = mag + ((size_t)b * H_N + h) * NF_PAD;
  float m = mrow[i0] * (1.0f - f) + mrow[i1] * f;
  _Float16* o = feat + idx * 4;
  o[0] = (_Float16)px[0]; o[1] = (_Float16)px[1];
  o[2] = (_Float16)px[2]; o[3] = (_Float16)m;
}

// ---------------------------------------------------------- conv + LIF fused
__device__ inline h4 ld_tap(const _Float16* __restrict__ feat, int b, int hh, int ww) {
  h4 z = {};
  if ((unsigned)hh >= (unsigned)H_N || (unsigned)ww >= (unsigned)W_N) return z;
  return *(const h4*)(feat + ((((size_t)b * H_N + hh) * W_N) + ww) * 4);
}

__global__ void k_conv_lif(const _Float16* __restrict__ feat,
                           const _Float16* __restrict__ WcT,
                           const float* __restrict__ conv_b,
                           float* __restrict__ V,
                           float* __restrict__ ssum_t) {
  int tile = blockIdx.x * 4 + (threadIdx.x >> 5);
  int lane = threadIdx.x & 31;
  int b  = tile / (H_N * 14);
  int r  = tile % (H_N * 14);
  int h  = r / 14, wt = r % 14;
  int g = lane >> 4, ln = lane & 15;
  int w = wt * 16 + ln; // A-fragment row = output pixel

  // im2col A fragments: K = (ky*3+kx)*4 + c, padded to 64.
  v16h a0 = {}, a1 = {};
  if (g == 0) {
    h4 p0 = ld_tap(feat, b, h - 1, w - 1); // K 0..3
    h4 p1 = ld_tap(feat, b, h - 1, w    ); // K 4..7
    h4 p2 = ld_tap(feat, b, h - 1, w + 1); // K 8..11
    h4 p3 = ld_tap(feat, b, h,     w - 1); // K 12..15
#pragma unroll
    for (int j = 0; j < 4; ++j) {
      a0[j] = p0[j]; a0[4 + j] = p1[j]; a0[8 + j] = p2[j]; a0[12 + j] = p3[j];
    }
    h4 q = ld_tap(feat, b, h + 1, w + 1); // K 32..35 (second k-step)
#pragma unroll
    for (int j = 0; j < 4; ++j) a1[j] = q[j];
  } else {
    h4 p0 = ld_tap(feat, b, h,     w    ); // K 16..19
    h4 p1 = ld_tap(feat, b, h,     w + 1); // K 20..23
    h4 p2 = ld_tap(feat, b, h + 1, w - 1); // K 24..27
    h4 p3 = ld_tap(feat, b, h + 1, w    ); // K 28..31
#pragma unroll
    for (int j = 0; j < 4; ++j) {
      a0[j] = p0[j]; a0[4 + j] = p1[j]; a0[8 + j] = p2[j]; a0[12 + j] = p3[j];
    }
    // a1 stays zero (K 48..63 are padding)
  }

  const _Float16* B0 = WcT + (size_t)ln * 64;        // channels 0..15
  const _Float16* B1 = WcT + (size_t)(16 + ln) * 64; // channels 16..31
  v16h b00 = *(const v16h*)(B0 + 16 * g);
  v16h b01 = *(const v16h*)(B0 + 32 + 16 * g);
  v16h b10 = *(const v16h*)(B1 + 16 * g);
  v16h b11 = *(const v16h*)(B1 + 32 + 16 * g);

  v8f acc0 = {}, acc1 = {};
  acc0 = __builtin_amdgcn_wmma_f32_16x16x32_f16(false, a0, false, b00, (short)0, acc0, false, false);
  acc0 = __builtin_amdgcn_wmma_f32_16x16x32_f16(false, a1, false, b01, (short)0, acc0, false, false);
  acc1 = __builtin_amdgcn_wmma_f32_16x16x32_f16(false, a0, false, b10, (short)0, acc1, false, false);
  acc1 = __builtin_amdgcn_wmma_f32_16x16x32_f16(false, a1, false, b11, (short)0, acc1, false, false);

  float cb0 = conv_b[ln], cb1 = conv_b[16 + ln];
  float s0 = 0.0f, s1 = 0.0f;
#pragma unroll
  for (int i = 0; i < 8; ++i) {
    int wp = wt * 16 + i + 8 * g; // D-fragment row = output pixel
    size_t base = ((((size_t)b * H_N + h) * W_N) + wp) * CH_N;
    float v0 = BETA_ * V[base + ln] + acc0[i] + cb0;
    float sp0 = (v0 - THRESH_ > 0.0f) ? 1.0f : 0.0f;
    V[base + ln] = v0 - sp0 * THRESH_;
    s0 += sp0;
    float v1 = BETA_ * V[base + 16 + ln] + acc1[i] + cb1;
    float sp1 = (v1 - THRESH_ > 0.0f) ? 1.0f : 0.0f;
    V[base + 16 + ln] = v1 - sp1 * THRESH_;
    s1 += sp1;
  }
  // lanes ln and ln+16 hold the two pixel-halves of the same channel
  float o0 = __shfl_xor(s0, 16, 32);
  float o1 = __shfl_xor(s1, 16, 32);
  if (g == 0) {
    atomicAdd(&ssum_t[b * CH_N + ln],      s0 + o0);
    atomicAdd(&ssum_t[b * CH_N + 16 + ln], s1 + o1);
  }
}

// --------------------------------------------------------------- head/output
__global__ void k_head(const float* __restrict__ ssum,
                       const float* __restrict__ head_w,
                       const float* __restrict__ head_b,
                       const float* __restrict__ energy,
                       float* __restrict__ out) {
  __shared__ float lg[T_N * B_N * OUT_N];
  __shared__ float red[256];
  int tid = threadIdx.x;
  const float invHW = 1.0f / (float)(H_N * W_N);
  for (int idx = tid; idx < T_N * B_N * OUT_N; idx += 256) {
    int o = idx % OUT_N;
    int tb = idx / OUT_N; // t*B + b
    float acc = head_b[o];
    const float* sr = ssum + (size_t)tb * CH_N;
    for (int c = 0; c < CH_N; ++c) acc += sr[c] * invHW * head_w[c * OUT_N + o];
    lg[idx] = acc;
    out[80 + idx] = acc; // logits_seq
  }
  __syncthreads();
  for (int idx = tid; idx < B_N * OUT_N; idx += 256) {
    float s = 0.0f;
    for (int t = 0; t < T_N; ++t) s += lg[t * B_N * OUT_N + idx];
    out[idx] = s * (1.0f / (float)T_N); // readout
  }
  float ps = 0.0f;
  for (int i = tid; i < T_N * B_N * CH_N; i += 256) ps += ssum[i];
  red[tid] = ps;
  __syncthreads();
  for (int s = 128; s > 0; s >>= 1) {
    if (tid < s) red[tid] += red[tid + s];
    __syncthreads();
  }
  if (tid == 0) {
    out[1360] = red[0] / ((float)T_N * B_N * H_N * W_N * CH_N);       // mean spike rate
    out[1361] = energy[0] / ((float)T_N * B_N * H_N * (float)NFREQ);  // mean spectral energy
  }
}

// ---------------------------------------------------------------------------
extern "C" void kernel_launch(void* const* d_in, const int* in_sizes, int n_in,
                              void* d_out, int out_size, void* d_ws, size_t ws_size,
                              hipStream_t stream) {
  const float* x_seq  = (const float*)d_in[0];
  const float* conv_w = (const float*)d_in[1];
  const float* conv_b = (const float*)d_in[2];
  const float* head_w = (const float*)d_in[3];
  const float* head_b = (const float*)d_in[4];
  float* out = (float*)d_out;

  char* ws = (char*)d_ws;
  size_t off = 0;
  auto alloc = [&](size_t bytes) -> char* {
    char* p = ws + off;
    off += (bytes + 255) & ~(size_t)255;
    return p;
  };
  float*    V    = (float*)   alloc((size_t)B_N * H_N * W_N * CH_N * 4); // membrane, L2-resident
  _Float16* feat = (_Float16*)alloc((size_t)B_N * H_N * W_N * 4 * 2);
  _Float16* gray = (_Float16*)alloc((size_t)B_N * H_N * W_N * 2);
  _Float16* Rt   = (_Float16*)alloc((size_t)B_N * 256 * K224 * 2);
  float*    mag  = (float*)   alloc((size_t)B_N * H_N * NF_PAD * 4);
  _Float16* Wt   = (_Float16*)alloc((size_t)256 * K224 * 2);
  _Float16* Fre  = (_Float16*)alloc((size_t)K224 * K224 * 2);
  _Float16* Fim  = (_Float16*)alloc((size_t)K224 * K224 * 2);
  _Float16* Fin  = (_Float16*)alloc((size_t)K224 * K224 * 2);
  _Float16* WcT  = (_Float16*)alloc((size_t)CH_N * 64 * 2);
  float*    ssum = (float*)   alloc((size_t)T_N * B_N * CH_N * 4);
  float*    energy = (float*) alloc(256);
  if (off > ws_size) return; // workspace too small; nothing safe to do

  // deterministic per-call init (harness does not re-poison between replays)
  {
    size_t nV = (size_t)B_N * H_N * W_N * CH_N;
    k_zero<<<(unsigned)((nV + 255) / 256), 256, 0, stream>>>(V, nV);
    size_t nS = (size_t)T_N * B_N * CH_N;
    k_zero<<<(unsigned)((nS + 255) / 256), 256, 0, stream>>>(ssum, nS);
    k_zero<<<1, 256, 0, stream>>>(energy, 1);
  }
  k_init_wt<<<(256 * K224 + 255) / 256, 256, 0, stream>>>(Wt);
  k_init_f<<<(K224 * K224 + 255) / 256, 256, 0, stream>>>(Fre, Fim, Fin);
  k_init_wct<<<(CH_N * 64 + 255) / 256, 256, 0, stream>>>(conv_w, WcT);

  const unsigned pixBlocks = (unsigned)(((size_t)B_N * H_N * W_N + 255) / 256); // 1568
  for (int t = 0; t < T_N; ++t) {
    const float* x_t = x_seq + (size_t)t * B_N * H_N * W_N * 3;
    k_gray  <<<pixBlocks, 256, 0, stream>>>(x_t, gray);
    k_rowdft<<<(B_N * 14 * 16) / 4, 128, 0, stream>>>(gray, Wt, Rt);          // 448 blocks
    k_coldft<<<(B_N * 14 * 8) / 4, 128, 0, stream>>>(Fre, Fim, Fin, Rt, mag, energy); // 224 blocks
    k_feat  <<<pixBlocks, 256, 0, stream>>>(x_t, mag, feat);
    k_conv_lif<<<(B_N * H_N * 14) / 4, 128, 0, stream>>>(feat, WcT, conv_b, V,
                                                         ssum + (size_t)t * B_N * CH_N); // 6272 blocks
  }
  k_head<<<1, 256, 0, stream>>>(ssum, head_w, head_b, energy, out);
}
